// PW_Linear_31310311588556
// MI455X (gfx1250) — matrix operand order repaired and checked by reading, not compiled, not run on hardware
//
#include <hip/hip_runtime.h>

// ---------------------------------------------------------------------------
// Types for CDNA5 WMMA (wave32, v_wmma_f32_16x16x32_bf16)
// ---------------------------------------------------------------------------
typedef __bf16 bf16_t;
typedef bf16_t v16bf  __attribute__((ext_vector_type(16)));
typedef bf16_t bf16x2 __attribute__((ext_vector_type(2)));
typedef float  v8f    __attribute__((ext_vector_type(8)));

union Frag  { v16bf v; uint4 q[2]; unsigned short s[16]; };
union BiasU { float4 f[2]; v8f v; };

#define WMMA_BF16(a, b, c) \
  __builtin_amdgcn_wmma_f32_16x16x32_bf16(false, (a).v, false, (b).v, (short)0, (c), false, false)

__device__ __forceinline__ unsigned short f2bf(float f) {
  __bf16 h = (__bf16)f;
  return __builtin_bit_cast(unsigned short, h);
}
// one v_cvt_pk_bf16_f32, no re-packing
__device__ __forceinline__ unsigned int pkbf(float a, float b) {
  bf16x2 p;
  p.x = (__bf16)a;
  p.y = (__bf16)b;
  return __builtin_bit_cast(unsigned int, p);
}

// Channel permutation: the packed C output of tile t (channels t*16+8h+v in
// lane (l15,h)) is exactly the B-fragment K = k*32+h*16+p*8+v, t = 2k+p.
// So the *next* GEMM's weight columns are permuted: col K <- channel n.
__device__ __host__ __forceinline__ int invperm(int K) {
  int k = K >> 5, h = (K >> 4) & 1, p = (K >> 3) & 1, v = K & 7;
  return ((2 * k + p) << 4) + (h << 3) + v;
}

// ---------------------------------------------------------------------------
// Prep: W1 zero-padded K=16->32 (natural), W2/W3 bf16 with permuted columns
// ---------------------------------------------------------------------------
__global__ void prep_bf16(const float* __restrict__ W1,
                          const float* __restrict__ W2,
                          const float* __restrict__ W3,
                          unsigned short* __restrict__ ws) {
  int i = blockIdx.x * blockDim.x + threadIdx.x;
  if (i < 8192) {                               // W1 padded: row n, col k(0..31)
    int row = i >> 5, col = i & 31;
    ws[i] = (col < 16) ? f2bf(W1[row * 16 + col]) : (unsigned short)0;
  } else if (i < 73728) {                       // W2, permuted columns
    int j = i - 8192;
    int row = j >> 8, K = j & 255;
    ws[i] = f2bf(W2[row * 256 + invperm(K)]);
  } else if (i < 204800) {                      // W3, permuted columns
    int j = i - 73728;
    int row = j >> 8, K = j & 255;
    ws[i] = f2bf(W3[row * 256 + invperm(K)]);
  }
}

// ---------------------------------------------------------------------------
// Fused kernel: one wave = 32 batch rows (two N=16 tiles share weight frags).
//   h^T = W @ x^T : A = weights (global, contiguous 16B loads)
//                   B = activations (lane-private LDS frag regions / VGPRs)
// Activations bounce through lane-private, bank-tiled frag regions; h2 frags
// are hoisted into 128 VGPRs so GEMM3 (2/3 of the FLOPs) does zero LDS reads.
// ---------------------------------------------------------------------------
constexpr int LST = 264;   // per-lane frag region (ushorts) = 528 B

__global__ __launch_bounds__(32)
void flow_kernel(const float* __restrict__ x,
                 const float* __restrict__ b1,
                 const float* __restrict__ b2,
                 const float* __restrict__ b3,
                 const unsigned short* __restrict__ w1p,   // 256x32 padded
                 const unsigned short* __restrict__ w2p,   // 256x256 perm cols
                 const unsigned short* __restrict__ w3p,   // 512x256 perm cols
                 float* __restrict__ out, int nrows) {
  __shared__ __attribute__((aligned(16))) unsigned short sf1[32 * LST];
  __shared__ __attribute__((aligned(16))) unsigned short sf2[32 * LST];
  __shared__ __attribute__((aligned(16))) float swn[32 * 32];
  __shared__ __attribute__((aligned(16))) float scum[32 * 33];

  const int lane = threadIdx.x & 31;
  const int half = lane >> 4;
  const int l15  = lane & 15;
  const int rowBase = blockIdx.x * 32;
  unsigned short* myf1 = sf1 + (size_t)lane * LST;
  unsigned short* myf2 = sf2 + (size_t)lane * LST;

  // ---- B fragments for GEMM1: x_c^T (even cols), K=16 padded to 32 ----
  Frag bx[2];
  const unsigned int msk = half ? 0u : 0xFFFFFFFFu;
#pragma unroll
  for (int mb = 0; mb < 2; ++mb) {
    const float4* xr = (const float4*)(x + (size_t)(rowBase + mb * 16 + l15) * 32);
    float4 f0 = xr[0], f1 = xr[1], f2 = xr[2], f3 = xr[3];
    float4 f4 = xr[4], f5 = xr[5], f6 = xr[6], f7 = xr[7];
    bx[mb].q[0] = make_uint4(pkbf(f0.x, f0.z) & msk, pkbf(f1.x, f1.z) & msk,
                             pkbf(f2.x, f2.z) & msk, pkbf(f3.x, f3.z) & msk);
    bx[mb].q[1] = make_uint4(pkbf(f4.x, f4.z) & msk, pkbf(f5.x, f5.z) & msk,
                             pkbf(f6.x, f6.z) & msk, pkbf(f7.x, f7.z) & msk);
  }

  // ---- GEMM1: h1^T = W1 @ x_c^T + b1 -> lane-private frag region ----
  for (int t = 0; t < 16; ++t) {
    Frag a;
    const unsigned short* wrow = w1p + (size_t)(t * 16 + l15) * 32;
    a.q[0] = *(const uint4*)(wrow + half * 16);
    a.q[1] = *(const uint4*)(wrow + half * 16 + 8);
    v8f c0 = {}, c1 = {};
    c0 = WMMA_BF16(a, bx[0], c0);
    c1 = WMMA_BF16(a, bx[1], c1);
    BiasU bu;
    bu.f[0] = *(const float4*)(b1 + t * 16 + 8 * half);
    bu.f[1] = *(const float4*)(b1 + t * 16 + 8 * half + 4);
    c0 = c0 + bu.v;
    c1 = c1 + bu.v;
    uint4 s0 = make_uint4(pkbf(c0[0], c0[1]), pkbf(c0[2], c0[3]),
                          pkbf(c0[4], c0[5]), pkbf(c0[6], c0[7]));
    uint4 s1 = make_uint4(pkbf(c1[0], c1[1]), pkbf(c1[2], c1[3]),
                          pkbf(c1[4], c1[5]), pkbf(c1[6], c1[7]));
    const int off = (t >> 1) * 16 + (t & 1) * 8;    // frag k = t>>1, dword-half = t&1
    *(uint4*)(myf1 +       off) = s0;               // mb = 0
    *(uint4*)(myf1 + 128 + off) = s1;               // mb = 1
  }

  // ---- GEMM2: h2^T = relu(W2p @ h1^T + b2) -> frag region ----
  const v8f vzero = {};
  for (int t = 0; t < 16; ++t) {
    v8f c0 = {}, c1 = {};
    const unsigned short* wrow = w2p + (size_t)(t * 16 + l15) * 256;
#pragma unroll
    for (int k = 0; k < 8; ++k) {
      Frag a, g0, g1;
      a.q[0] = *(const uint4*)(wrow + k * 32 + half * 16);
      a.q[1] = *(const uint4*)(wrow + k * 32 + half * 16 + 8);
      g0.q[0] = *(const uint4*)(myf1 +       k * 16);
      g0.q[1] = *(const uint4*)(myf1 +       k * 16 + 8);
      g1.q[0] = *(const uint4*)(myf1 + 128 + k * 16);
      g1.q[1] = *(const uint4*)(myf1 + 128 + k * 16 + 8);
      c0 = WMMA_BF16(a, g0, c0);
      c1 = WMMA_BF16(a, g1, c1);
    }
    BiasU bu;
    bu.f[0] = *(const float4*)(b2 + t * 16 + 8 * half);
    bu.f[1] = *(const float4*)(b2 + t * 16 + 8 * half + 4);
    c0 = __builtin_elementwise_max(c0 + bu.v, vzero);
    c1 = __builtin_elementwise_max(c1 + bu.v, vzero);
    uint4 s0 = make_uint4(pkbf(c0[0], c0[1]), pkbf(c0[2], c0[3]),
                          pkbf(c0[4], c0[5]), pkbf(c0[6], c0[7]));
    uint4 s1 = make_uint4(pkbf(c1[0], c1[1]), pkbf(c1[2], c1[3]),
                          pkbf(c1[4], c1[5]), pkbf(c1[6], c1[7]));
    const int off = (t >> 1) * 16 + (t & 1) * 8;
    *(uint4*)(myf2 +       off) = s0;
    *(uint4*)(myf2 + 128 + off) = s1;
  }

  // ---- hoist h2 fragments into registers: GEMM3 does zero LDS reads ----
  Frag hf0[8], hf1[8];
#pragma unroll
  for (int k = 0; k < 8; ++k) {
    hf0[k].q[0] = *(const uint4*)(myf2 +       k * 16);
    hf0[k].q[1] = *(const uint4*)(myf2 +       k * 16 + 8);
    hf1[k].q[0] = *(const uint4*)(myf2 + 128 + k * 16);
    hf1[k].q[1] = *(const uint4*)(myf2 + 128 + k * 16 + 8);
  }

  // ---- GEMM3: 512 logits per row, online softmax stats ----
  float mrun0 = -3.0e38f, srun0 = 0.f, mrun1 = -3.0e38f, srun1 = 0.f;
  v8f keepA0 = {}, keepA1 = {}, keepB0 = {}, keepB1 = {};   // logit cols 0..31

  for (int t = 0; t < 32; ++t) {
    v8f c0 = {}, c1 = {};
    const unsigned short* wrow = w3p + (size_t)(t * 16 + l15) * 256;
#pragma unroll
    for (int k = 0; k < 8; ++k) {
      Frag a;
      a.q[0] = *(const uint4*)(wrow + k * 32 + half * 16);
      a.q[1] = *(const uint4*)(wrow + k * 32 + half * 16 + 8);
      c0 = WMMA_BF16(a, hf0[k], c0);
      c1 = WMMA_BF16(a, hf1[k], c1);
    }
    BiasU bu;
    bu.f[0] = *(const float4*)(b3 + t * 16 + 8 * half);
    bu.f[1] = *(const float4*)(b3 + t * 16 + 8 * half + 4);
    c0 = c0 + bu.v;
    c1 = c1 + bu.v;
    if (t == 0) { keepA0 = c0; keepA1 = c1; }
    if (t == 1) { keepB0 = c0; keepB1 = c1; }

    // Online max/sum-exp; each row's logits live in lanes (m, m+16) -> one xor-16.
    {
      float mx = c0[0];
#pragma unroll
      for (int v = 1; v < 8; ++v) mx = fmaxf(mx, c0[v]);
      mx = fmaxf(mx, __shfl_xor(mx, 16, 32));
      float nm = fmaxf(mrun0, mx);
      float e = 0.f;
#pragma unroll
      for (int v = 0; v < 8; ++v) e += __expf(c0[v] - nm);
      e += __shfl_xor(e, 16, 32);
      srun0 = srun0 * __expf(mrun0 - nm) + e;
      mrun0 = nm;
    }
    {
      float mx = c1[0];
#pragma unroll
      for (int v = 1; v < 8; ++v) mx = fmaxf(mx, c1[v]);
      mx = fmaxf(mx, __shfl_xor(mx, 16, 32));
      float nm = fmaxf(mrun1, mx);
      float e = 0.f;
#pragma unroll
      for (int v = 0; v < 8; ++v) e += __expf(c1[v] - nm);
      e += __shfl_xor(e, 16, 32);
      srun1 = srun1 * __expf(mrun1 - nm) + e;
      mrun1 = nm;
    }
  }

  // ---- normalized W_n columns 0..31 (the only gathered ones) -> LDS ----
  {
    float inv0 = 1.0f / srun0, inv1 = 1.0f / srun1;
    float* wr0 = swn + (size_t)l15 * 32;
    float* wr1 = swn + (size_t)(16 + l15) * 32;
#pragma unroll
    for (int g = 0; g < 2; ++g) {
      const v8f& k0 = g ? keepB0 : keepA0;
      const v8f& k1 = g ? keepB1 : keepA1;
      const int off = g * 16 + 8 * half;
      *(float4*)(wr0 + off) = make_float4(
          __expf(k0[0] - mrun0) * inv0, __expf(k0[1] - mrun0) * inv0,
          __expf(k0[2] - mrun0) * inv0, __expf(k0[3] - mrun0) * inv0);
      *(float4*)(wr0 + off + 4) = make_float4(
          __expf(k0[4] - mrun0) * inv0, __expf(k0[5] - mrun0) * inv0,
          __expf(k0[6] - mrun0) * inv0, __expf(k0[7] - mrun0) * inv0);
      *(float4*)(wr1 + off) = make_float4(
          __expf(k1[0] - mrun1) * inv1, __expf(k1[1] - mrun1) * inv1,
          __expf(k1[2] - mrun1) * inv1, __expf(k1[3] - mrun1) * inv1);
      *(float4*)(wr1 + off + 4) = make_float4(
          __expf(k1[4] - mrun1) * inv1, __expf(k1[5] - mrun1) * inv1,
          __expf(k1[6] - mrun1) * inv1, __expf(k1[7] - mrun1) * inv1);
    }
  }

  // ---- epilogue: one lane per batch row (all 32 lanes active) ----
  {
    const int r = lane;
    const int row = rowBase + r;
    float csum = 0.f;
#pragma unroll 4
    for (int jc = 0; jc < 32; ++jc) {
      scum[r * 33 + jc] = csum;
      csum += swn[r * 32 + jc];
    }
    scum[r * 33 + 32] = csum;

    const float2* xr = (const float2*)(x + (size_t)row * 32);
    float2* zr = (float2*)(out + (size_t)row * 32);
    float ld = 0.f;
    for (int j = 0; j < 16; ++j) {
      float2 xv = xr[j];                      // (x_c[j], x_t[j])
      float tq = xv.y * 32.0f;
      int bq = (int)tq;                       // floor, x_t in [0,1)
      float alpha = tq - (float)bq;
      float g  = swn[r * 32 + bq];
      float gc = scum[r * 33 + bq];
      float2 o;
      o.x = xv.x;
      o.y = alpha * g + gc;
      zr[j] = o;
      ld += __logf(g * 32.0f);
    }
    out[(size_t)nrows * 32 + row] = ld;       // log|det J|
  }
}

// ---------------------------------------------------------------------------
// Launch
// ---------------------------------------------------------------------------
extern "C" void kernel_launch(void* const* d_in, const int* in_sizes, int n_in,
                              void* d_out, int out_size, void* d_ws, size_t ws_size,
                              hipStream_t stream) {
  const float* x  = (const float*)d_in[0];
  const float* W1 = (const float*)d_in[1];
  const float* b1 = (const float*)d_in[2];
  const float* W2 = (const float*)d_in[3];
  const float* b2 = (const float*)d_in[4];
  const float* W3 = (const float*)d_in[5];
  const float* b3 = (const float*)d_in[6];
  float* out = (float*)d_out;
  unsigned short* ws = (unsigned short*)d_ws;

  const int nrows = in_sizes[0] / 32;          // B = 131072
  const int NCONV = 8192 + 65536 + 131072;     // padded W1 + W2 + W3
  prep_bf16<<<(NCONV + 255) / 256, 256, 0, stream>>>(W1, W2, W3, ws);

  unsigned short* w1p = ws;
  unsigned short* w2p = ws + 8192;
  unsigned short* w3p = ws + 8192 + 65536;

  const int blocks = nrows / 32;               // 32 batch rows per wave/block
  flow_kernel<<<blocks, 32, 0, stream>>>(x, b1, b2, b3, w1p, w2p, w3p, out, nrows);
  (void)n_in; (void)out_size; (void)ws_size;
}
